// ThisGPT2Attention_34909494181925
// MI455X (gfx1250) — matrix operand order repaired and checked by reading, not compiled, not run on hardware
//
#include <hip/hip_runtime.h>

// ---------------------------------------------------------------------------
// GPT-2 attention for MI455X (gfx1250, wave32, WMMA).
// B=2, S=2048, D=1024, H=16, hd=64.
// All matrix math via v_wmma_f32_16x16x32_f16 (fp32 accumulation).
// GEMMs: 64x64 register-blocked wave tiles (4x4 WMMA frags, 4x data reuse).
// Attention: flash-style, 32 q-rows per wave (2x K/V reuse).
// ---------------------------------------------------------------------------

typedef __attribute__((ext_vector_type(16))) _Float16 v16h;
typedef __attribute__((ext_vector_type(8)))  float    v8f;

union FragH { v16h v; _Float16 h[16]; unsigned u[8]; };

#define BB   2
#define SS   2048
#define DD   1024
#define HH   16
#define HDD  64
#define MM   (BB * SS)        // 4096 rows
#define NQKV (3 * DD)         // 3072

// ---- WMMA fragment loaders (layouts per CDNA5 ISA 7.12.2) ------------------
// A fragment: 16x32 f16, element (m, k); source row-major, base at tile (0,0).
__device__ __forceinline__ v16h load_a16(const _Float16* __restrict__ base,
                                         int lda, int lane) {
  int m = lane & 15, kh = (lane >> 4) & 1;
  const unsigned* row = (const unsigned*)(base + (size_t)m * lda);
  FragH f;
#pragma unroll
  for (int j = 0; j < 8; ++j) {
    int k = ((j & 4) ? 16 : 0) + kh * 8 + 2 * (j & 3);
    f.u[j] = row[k >> 1];          // two consecutive f16 (k, k+1)
  }
  return f.v;
}

// B fragment: 32x16 f16, element (k, n); storage K-contiguous per column n:
// element (k,n) at base[n*ldb + k].
__device__ __forceinline__ v16h load_bt16(const _Float16* __restrict__ base,
                                          int ldb, int lane) {
  int n = lane & 15, kh = (lane >> 4) & 1;
  const unsigned* col = (const unsigned*)(base + (size_t)n * ldb);
  FragH f;
#pragma unroll
  for (int j = 0; j < 8; ++j) {
    int k = ((j & 4) ? 16 : 0) + kh * 8 + 2 * (j & 3);
    f.u[j] = col[k >> 1];
  }
  return f.v;
}

static __device__ __forceinline__ v8f wmma_f16(v16h a, v16h b, v8f c) {
  return __builtin_amdgcn_wmma_f32_16x16x32_f16(false, a, false, b,
                                                (short)0, c, false, false);
}

#define ZERO8 ((v8f){0.f,0.f,0.f,0.f,0.f,0.f,0.f,0.f})

// ---- Conversion / transpose kernels ---------------------------------------
__global__ __launch_bounds__(256) void cvt_f32_to_f16(const float* __restrict__ in,
                                                      _Float16* __restrict__ out,
                                                      int n) {
  int i = blockIdx.x * 256 + threadIdx.x;
  if (i < n) out[i] = (_Float16)in[i];
}

// in: [K][N] fp32 row-major  ->  out: [N][K] f16 (B frags read K-contiguous)
__global__ __launch_bounds__(256) void transpose_to_f16(const float* __restrict__ in,
                                                        _Float16* __restrict__ out,
                                                        int K, int N) {
  int i = blockIdx.x * 256 + threadIdx.x;
  if (i < K * N) {
    int k = i / N, n = i - k * N;
    out[(size_t)n * K + k] = (_Float16)in[i];
  }
}

// ---- QKV GEMM: [4096,1024] x [1024,3072] + bias -> scatter Q/K/Vt ---------
// 64x64 tile per wave: 4x4 C frags, 4x A/B reuse.
__global__ __launch_bounds__(256) void gemm_qkv(const _Float16* __restrict__ A,
                                                const _Float16* __restrict__ Wt,
                                                const float* __restrict__ bias,
                                                _Float16* __restrict__ Qb,
                                                _Float16* __restrict__ Kb,
                                                _Float16* __restrict__ Vt) {
  const int K = DD, NT = NQKV / 64;          // 48 column super-tiles
  int wave = threadIdx.x >> 5, lane = threadIdx.x & 31;
  int tile = blockIdx.x * 8 + wave;          // 3072 total
  int tm = tile / NT, tn = tile - tm * NT;

  const _Float16* Ab = A  + (size_t)(tm * 64) * K;
  const _Float16* Bb = Wt + (size_t)(tn * 64) * K;

  v8f acc[4][4];
#pragma unroll
  for (int ms = 0; ms < 4; ++ms)
#pragma unroll
    for (int ns = 0; ns < 4; ++ns) acc[ms][ns] = ZERO8;

#pragma unroll 2
  for (int kb = 0; kb < K; kb += 32) {
    v16h af[4];
#pragma unroll
    for (int ms = 0; ms < 4; ++ms)
      af[ms] = load_a16(Ab + (size_t)(ms * 16) * K + kb, K, lane);
#pragma unroll
    for (int ns = 0; ns < 4; ++ns) {
      v16h bf = load_bt16(Bb + (size_t)(ns * 16) * K + kb, K, lane);
#pragma unroll
      for (int ms = 0; ms < 4; ++ms)
        acc[ms][ns] = wmma_f16(af[ms], bf, acc[ms][ns]);
    }
  }

  int nl = lane & 15, half = lane >> 4;
#pragma unroll
  for (int ns = 0; ns < 4; ++ns) {
    int n = tn * 64 + ns * 16 + nl;
    float bn = bias[n];
    int which = n / DD;                      // 0=Q 1=K 2=V
    int nq = n - which * DD;
    int h = nq / HDD, col = nq - h * HDD;
#pragma unroll
    for (int ms = 0; ms < 4; ++ms) {
#pragma unroll
      for (int r = 0; r < 8; ++r) {
        int m = tm * 64 + ms * 16 + half * 8 + r;
        int bidx = m >> 11, s = m & (SS - 1);
        _Float16 hv = (_Float16)(acc[ms][ns][r] + bn);
        size_t bh = (size_t)(bidx * HH + h);
        if (which == 0)      Qb[(bh * SS + s) * HDD + col] = hv;
        else if (which == 1) Kb[(bh * SS + s) * HDD + col] = hv;
        else                 Vt[(bh * HDD + col) * SS + s] = hv;  // V transposed
      }
    }
  }
}

// ---- Flash attention: 32 q-rows per wave, 32 keys / iteration -------------
__global__ __launch_bounds__(256) void attn_kernel(const _Float16* __restrict__ Qb,
                                                   const _Float16* __restrict__ Kb,
                                                   const _Float16* __restrict__ Vt,
                                                   const float* __restrict__ mask,
                                                   _Float16* __restrict__ Ctx) {
  __shared__ _Float16 Plds[8][2][16][32];    // per-wave, per-qtile P relayout

  int wave = threadIdx.x >> 5, lane = threadIdx.x & 31;
  int nl = lane & 15, half = lane >> 4;
  int bh = blockIdx.x >> 3;                  // 32 (b,h) pairs
  int chunk = blockIdx.x & 7;                // 8 q-chunks of 256 rows
  int qbase = chunk * 256 + wave * 32;       // 32 q-rows per wave
  int b = bh >> 4, h = bh & 15;

  const _Float16* Q  = Qb + ((size_t)bh * SS + qbase) * HDD;
  const _Float16* Kp = Kb + (size_t)bh * SS * HDD;
  const _Float16* Vp = Vt + (size_t)bh * HDD * SS;
  const float* mrow  = mask + (size_t)b * SS;

  // Q fragments (2 q-tiles x 2 kd halves), pre-scaled by 1/sqrt(64)=0.125
  FragH qa[2][2];
#pragma unroll
  for (int qt = 0; qt < 2; ++qt) {
    qa[qt][0].v = load_a16(Q + (size_t)(qt * 16) * HDD,      HDD, lane);
    qa[qt][1].v = load_a16(Q + (size_t)(qt * 16) * HDD + 32, HDD, lane);
#pragma unroll
    for (int e = 0; e < 16; ++e) {
      qa[qt][0].h[e] = qa[qt][0].h[e] * (_Float16)0.125f;
      qa[qt][1].h[e] = qa[qt][1].h[e] * (_Float16)0.125f;
    }
  }

  float mrun[2][8], lrun[2][8];
  v8f acc[2][4];
#pragma unroll
  for (int qt = 0; qt < 2; ++qt) {
#pragma unroll
    for (int r = 0; r < 8; ++r) { mrun[qt][r] = -1e30f; lrun[qt][r] = 0.f; }
#pragma unroll
    for (int t = 0; t < 4; ++t) acc[qt][t] = ZERO8;
  }

  for (int kb = 0; kb < SS; kb += 32) {
    // ---- scores: 2 key tiles x 2 q tiles; hd=64 -> 2 chained WMMAs each ----
    v8f sc[2][2];                             // [qt][key tile]
#pragma unroll
    for (int qt = 0; qt < 2; ++qt) { sc[qt][0] = ZERO8; sc[qt][1] = ZERO8; }
    {
      v16h b00 = load_bt16(Kp + (size_t)kb * HDD,             HDD, lane);
      v16h b01 = load_bt16(Kp + (size_t)kb * HDD + 32,        HDD, lane);
#pragma unroll
      for (int qt = 0; qt < 2; ++qt) {
        sc[qt][0] = wmma_f16(qa[qt][0].v, b00, sc[qt][0]);
        sc[qt][0] = wmma_f16(qa[qt][1].v, b01, sc[qt][0]);
      }
      v16h b10 = load_bt16(Kp + (size_t)(kb + 16) * HDD,      HDD, lane);
      v16h b11 = load_bt16(Kp + (size_t)(kb + 16) * HDD + 32, HDD, lane);
#pragma unroll
      for (int qt = 0; qt < 2; ++qt) {
        sc[qt][1] = wmma_f16(qa[qt][0].v, b10, sc[qt][1]);
        sc[qt][1] = wmma_f16(qa[qt][1].v, b11, sc[qt][1]);
      }
    }

    // ---- online softmax over the 32 keys, per q tile ----
    float mv0 = mrow[kb + nl], mv1 = mrow[kb + 16 + nl];
    float corr[2][8];
#pragma unroll
    for (int qt = 0; qt < 2; ++qt) {
#pragma unroll
      for (int r = 0; r < 8; ++r) {
        float s0 = sc[qt][0][r] + mv0;
        float s1 = sc[qt][1][r] + mv1;
        float mx = fmaxf(s0, s1);
#pragma unroll
        for (int off = 1; off < 16; off <<= 1)
          mx = fmaxf(mx, __shfl_xor(mx, off, 32));   // row reduce in half-wave
        float nm = fmaxf(mrun[qt][r], mx);
        corr[qt][r] = __expf(mrun[qt][r] - nm);
        float p0 = __expf(s0 - nm);
        float p1 = __expf(s1 - nm);
        float ps = p0 + p1;
#pragma unroll
        for (int off = 1; off < 16; off <<= 1)
          ps += __shfl_xor(ps, off, 32);
        lrun[qt][r] = lrun[qt][r] * corr[qt][r] + ps;
        mrun[qt][r] = nm;
        Plds[wave][qt][half * 8 + r][nl]      = (_Float16)p0;
        Plds[wave][qt][half * 8 + r][16 + nl] = (_Float16)p1;
      }
    }
    __builtin_amdgcn_wave_barrier();       // intra-wave LDS store->load ordering

    v16h pa[2];
#pragma unroll
    for (int qt = 0; qt < 2; ++qt)
      pa[qt] = load_a16(&Plds[wave][qt][0][0], 32, lane);

    // ---- ctx += P(16x32) x V(32x64): V frags shared by both q tiles ----
#pragma unroll
    for (int qt = 0; qt < 2; ++qt)
#pragma unroll
      for (int t = 0; t < 4; ++t)
#pragma unroll
        for (int r = 0; r < 8; ++r) acc[qt][t][r] *= corr[qt][r];
#pragma unroll
    for (int t = 0; t < 4; ++t) {
      v16h vb = load_bt16(Vp + (size_t)(t * 16) * SS + kb, SS, lane);
#pragma unroll
      for (int qt = 0; qt < 2; ++qt)
        acc[qt][t] = wmma_f16(pa[qt], vb, acc[qt][t]);
    }
  }

  // ---- normalize and store ctx as f16 [B,S,D] ----
#pragma unroll
  for (int qt = 0; qt < 2; ++qt) {
    float inv[8];
#pragma unroll
    for (int r = 0; r < 8; ++r) inv[r] = 1.0f / lrun[qt][r];
#pragma unroll
    for (int t = 0; t < 4; ++t) {
#pragma unroll
      for (int r = 0; r < 8; ++r) {
        int s = qbase + qt * 16 + half * 8 + r;
        Ctx[((size_t)(b * SS + s)) * DD + h * HDD + t * 16 + nl] =
            (_Float16)(acc[qt][t][r] * inv[r]);
      }
    }
  }
}

// ---- Output projection: [4096,1024] x [1024,1024] + bias -> fp32 ----------
__global__ __launch_bounds__(256) void gemm_proj(const _Float16* __restrict__ A,
                                                 const _Float16* __restrict__ Wt,
                                                 const float* __restrict__ bias,
                                                 float* __restrict__ Out) {
  const int K = DD, NT = DD / 64;            // 16 column super-tiles
  int wave = threadIdx.x >> 5, lane = threadIdx.x & 31;
  int tile = blockIdx.x * 8 + wave;          // 1024 total
  int tm = tile / NT, tn = tile - tm * NT;

  const _Float16* Ab = A  + (size_t)(tm * 64) * K;
  const _Float16* Bb = Wt + (size_t)(tn * 64) * K;

  v8f acc[4][4];
#pragma unroll
  for (int ms = 0; ms < 4; ++ms)
#pragma unroll
    for (int ns = 0; ns < 4; ++ns) acc[ms][ns] = ZERO8;

#pragma unroll 2
  for (int kb = 0; kb < K; kb += 32) {
    v16h af[4];
#pragma unroll
    for (int ms = 0; ms < 4; ++ms)
      af[ms] = load_a16(Ab + (size_t)(ms * 16) * K + kb, K, lane);
#pragma unroll
    for (int ns = 0; ns < 4; ++ns) {
      v16h bf = load_bt16(Bb + (size_t)(ns * 16) * K + kb, K, lane);
#pragma unroll
      for (int ms = 0; ms < 4; ++ms)
        acc[ms][ns] = wmma_f16(af[ms], bf, acc[ms][ns]);
    }
  }

  int nl = lane & 15, half = lane >> 4;
#pragma unroll
  for (int ns = 0; ns < 4; ++ns) {
    int n = tn * 64 + ns * 16 + nl;
    float bn = bias[n];
#pragma unroll
    for (int ms = 0; ms < 4; ++ms) {
#pragma unroll
      for (int r = 0; r < 8; ++r) {
        int m = tm * 64 + ms * 16 + half * 8 + r;
        Out[(size_t)m * DD + n] = acc[ms][ns][r] + bn;
      }
    }
  }
}

// ---------------------------------------------------------------------------
extern "C" void kernel_launch(void* const* d_in, const int* in_sizes, int n_in,
                              void* d_out, int out_size, void* d_ws, size_t ws_size,
                              hipStream_t stream) {
  const float* hidden = (const float*)d_in[0];
  const float* mask   = (const float*)d_in[1];
  const float* Wqkv   = (const float*)d_in[2];
  const float* bqkv   = (const float*)d_in[3];
  const float* Wproj  = (const float*)d_in[4];
  const float* bproj  = (const float*)d_in[5];
  float* out = (float*)d_out;

  char* ws = (char*)d_ws;
  size_t off = 0;
  _Float16* hid_h   = (_Float16*)(ws + off); off += (size_t)MM * DD * 2;    // 8 MB
  _Float16* wqkv_t  = (_Float16*)(ws + off); off += (size_t)DD * NQKV * 2;  // 6 MB
  _Float16* wproj_t = (_Float16*)(ws + off); off += (size_t)DD * DD * 2;    // 2 MB
  _Float16* Qb      = (_Float16*)(ws + off); off += (size_t)MM * DD * 2;    // 8 MB
  _Float16* Kb      = (_Float16*)(ws + off); off += (size_t)MM * DD * 2;    // 8 MB
  _Float16* Vt      = (_Float16*)(ws + off); off += (size_t)MM * DD * 2;    // 8 MB
  _Float16* Ctx     = (_Float16*)(ws + off); off += (size_t)MM * DD * 2;    // 8 MB

  cvt_f32_to_f16<<<(MM * DD) / 256, 256, 0, stream>>>(hidden, hid_h, MM * DD);
  transpose_to_f16<<<(DD * NQKV) / 256, 256, 0, stream>>>(Wqkv, wqkv_t, DD, NQKV);
  transpose_to_f16<<<(DD * DD) / 256, 256, 0, stream>>>(Wproj, wproj_t, DD, DD);

  // (4096/64)*(3072/64) = 3072 wave-tiles, 8 waves/block -> 384 blocks
  gemm_qkv<<<384, 256, 0, stream>>>(hid_h, wqkv_t, bqkv, Qb, Kb, Vt);

  // 32 (b,h) * 8 q-chunks -> 256 blocks, 8 waves x 32 q-rows each
  attn_kernel<<<256, 256, 0, stream>>>(Qb, Kb, Vt, mask, Ctx);

  // (4096/64)*(1024/64) = 1024 wave-tiles -> 128 blocks
  gemm_proj<<<128, 256, 0, stream>>>(Ctx, wproj_t, bproj, out);
}
